// SiglipAttention_7696581394904
// MI455X (gfx1250) — compile-verified, matrix operand order checked by reading, not run on hardware
//
#include <hip/hip_runtime.h>
#include <stdint.h>
#include <math.h>

typedef __attribute__((ext_vector_type(16))) __bf16 v16bf;
typedef __attribute__((ext_vector_type(8)))  float  v8f;

#define BB 8
#define SS 1024
#define EE 1152
#define HH 16
#define DD 72
#define DPQK 96          // padded head dim for Q/K (multiple of 32 for WMMA K-steps)
#define DPV  80          // padded head dim for V / O accumulators (5 N-tiles of 16)
#define MTOT (BB*SS)     // 8192 rows

#define MODE_PAD 0
#define MODE_VT  1
#define MODE_F32 2

union FragB { v16bf v; unsigned int d[8]; unsigned short u[16]; };

__device__ __forceinline__ unsigned short f2bf(float f){
  unsigned int u = __float_as_uint(f);
  unsigned int r = u + 0x7FFFu + ((u >> 16) & 1u);   // round-to-nearest-even
  return (unsigned short)(r >> 16);
}
__device__ __forceinline__ float bf2f(unsigned short h){
  return __uint_as_float(((unsigned int)h) << 16);
}
// dword index (within a 16-dword / 32-element row chunk) for bf16 A/B fragment VGPR r,
// lane-half h (lane>=16): VGPRs 0-3 -> K pairs {0..7}+8h ; VGPRs 4-7 -> {16..23}+8h
__device__ __forceinline__ int fragDw(int r, int h){ return (r < 4 ? r : r + 4) + 4*h; }

// ---------------- elementwise helpers ----------------
__global__ void k_f32_to_bf16(const float* __restrict__ in, unsigned short* __restrict__ out, int n){
  int i = blockIdx.x*256 + threadIdx.x;
  if (i < n) out[i] = f2bf(in[i]);
}
__global__ void k_zero_u32(unsigned int* __restrict__ p, int n){
  int i = blockIdx.x*256 + threadIdx.x;
  if (i < n) p[i] = 0u;
}

// ---------------- RoPE (in place on padded [BH][S][96] bf16) ----------------
__global__ void k_rope(unsigned short* __restrict__ X, float scale, int total){
  int t = blockIdx.x*256 + threadIdx.x;
  if (t >= total) return;
  int i  = t % 36;                 // pair index (d, d+36)
  int s  = (t / 36) & (SS-1);
  int bh = t / (36*SS);
  size_t base = ((size_t)bh*SS + s)*DPQK;
  float x0 = bf2f(X[base + i]);
  float x1 = bf2f(X[base + i + 36]);
  // inv_freq = 10000^(-i/36)
  float ang = (float)s * __expf(-(float)i * (9.210340371976184f / 36.0f));
  float cv, sv; sincosf(ang, &cv, &sv);
  X[base + i]      = f2bf((x0*cv - x1*sv) * scale);
  X[base + i + 36] = f2bf((x1*cv + x0*sv) * scale);
}

// ---------------- register-blocked WMMA GEMM ----------------
// Y[m,n] = sum_k A[m,k]*W[n,k] + bias[n]
// One wave computes a 64x64 tile: 16 C-fragments, 16 WMMAs per 32-wide K-step.
// Per K-step loads 4 A-frags + 4 B-frags (8KB/wave) for 256K FLOP -> ~32 FLOP/B.
__global__ __launch_bounds__(32) void k_gemm(const unsigned short* __restrict__ A,
                                             const unsigned short* __restrict__ W,
                                             const float* __restrict__ bias,
                                             void* __restrict__ dst, int mode){
  const int L  = threadIdx.x;
  const int lm = L & 15;
  const int h2 = L >> 4;
  const int m0 = blockIdx.x * 64;
  const int n0 = blockIdx.y * 64;

  const unsigned int* arow[4];
  const unsigned int* wrow[4];
  #pragma unroll
  for (int i = 0; i < 4; ++i){
    arow[i] = (const unsigned int*)(A + (size_t)(m0 + i*16 + lm) * EE);
    wrow[i] = (const unsigned int*)(W + (size_t)(n0 + i*16 + lm) * EE);
  }

  const v8f vz = {0.f,0.f,0.f,0.f,0.f,0.f,0.f,0.f};
  v8f c[4][4];
  #pragma unroll
  for (int mi = 0; mi < 4; ++mi)
    #pragma unroll
    for (int ni = 0; ni < 4; ++ni) c[mi][ni] = vz;

  for (int kc = 0; kc < EE/32; ++kc){
    FragB a[4];
    #pragma unroll
    for (int mi = 0; mi < 4; ++mi)
      #pragma unroll
      for (int r = 0; r < 8; ++r)
        a[mi].d[r] = arow[mi][kc*16 + fragDw(r,h2)];
    #pragma unroll
    for (int ni = 0; ni < 4; ++ni){
      FragB b;
      #pragma unroll
      for (int r = 0; r < 8; ++r)
        b.d[r] = wrow[ni][kc*16 + fragDw(r,h2)];
      #pragma unroll
      for (int mi = 0; mi < 4; ++mi)
        c[mi][ni] = __builtin_amdgcn_wmma_f32_16x16x32_bf16(false, a[mi].v, false, b.v,
                                                            (short)0, c[mi][ni], false, false);
    }
  }

  // epilogue
  #pragma unroll
  for (int ni = 0; ni < 4; ++ni){
    const int f  = n0 + ni*16 + lm;
    const float bv = bias[f];
    if (mode == MODE_F32){
      float* out = (float*)dst;
      #pragma unroll
      for (int mi = 0; mi < 4; ++mi)
        #pragma unroll
        for (int r = 0; r < 8; ++r){
          int row = m0 + mi*16 + r + 8*h2;
          out[(size_t)row*EE + f] = c[mi][ni][r] + bv;
        }
    } else {
      unsigned short* out = (unsigned short*)dst;
      int h = f / DD, d = f % DD;
      #pragma unroll
      for (int mi = 0; mi < 4; ++mi)
        #pragma unroll
        for (int r = 0; r < 8; ++r){
          int row = m0 + mi*16 + r + 8*h2;
          int b_  = row >> 10, s = row & (SS-1);
          float y = c[mi][ni][r] + bv;
          if (mode == MODE_PAD)
            out[((size_t)((b_*HH + h)*SS) + s)*DPQK + d] = f2bf(y);   // Q/K: [B,H,S,96]
          else
            out[((size_t)(b_*HH + h)*DPV + d)*SS + s]    = f2bf(y);   // V^T: [B,H,80,S]
        }
    }
  }
}

// ---------------- flash attention: one wave per (b,h,16-query tile) ----------------
__global__ __launch_bounds__(32) void k_attn(const unsigned short* __restrict__ Q,
                                             const unsigned short* __restrict__ K,
                                             const unsigned short* __restrict__ V,
                                             unsigned short* __restrict__ AO){
  __shared__ unsigned short pTile[16*32];   // P transpose staging (C-layout -> A-layout)
  const int L  = threadIdx.x;
  const int lm = L & 15, h2 = L >> 4;
  const int qt = blockIdx.x & 63;
  const int bh = blockIdx.x >> 6;
  const int b  = bh >> 4, h = bh & 15;
  const int s0 = qt * 16;

  // Q fragments (scale already folded in by RoPE kernel): 3 x (16x32)
  const unsigned int* qrow = (const unsigned int*)(Q + ((size_t)bh*SS + s0 + lm)*DPQK);
  FragB qf[3];
  #pragma unroll
  for (int c = 0; c < 3; ++c)
    #pragma unroll
    for (int r = 0; r < 8; ++r)
      qf[c].d[r] = qrow[c*16 + fragDw(r,h2)];

  v8f acc[5];
  const v8f vz = {0.f,0.f,0.f,0.f,0.f,0.f,0.f,0.f};
  #pragma unroll
  for (int t = 0; t < 5; ++t) acc[t] = vz;
  float rowm[8], rowl[8];
  #pragma unroll
  for (int r = 0; r < 8; ++r){ rowm[r] = -3.0e38f; rowl[r] = 0.f; }

  for (int kk = 0; kk < SS; kk += 32){
    // scores: 16 q x 32 keys, reduce padded D=96 in 3 WMMA K-steps
    v8f sc0 = vz, sc1 = vz;
    const unsigned int* kr0 = (const unsigned int*)(K + ((size_t)bh*SS + kk      + lm)*DPQK);
    const unsigned int* kr1 = (const unsigned int*)(K + ((size_t)bh*SS + kk + 16 + lm)*DPQK);
    #pragma unroll
    for (int c = 0; c < 3; ++c){
      FragB k0, k1;
      #pragma unroll
      for (int r = 0; r < 8; ++r){
        k0.d[r] = kr0[c*16 + fragDw(r,h2)];
        k1.d[r] = kr1[c*16 + fragDw(r,h2)];
      }
      sc0 = __builtin_amdgcn_wmma_f32_16x16x32_bf16(false, qf[c].v, false, k0.v, (short)0, sc0, false, false);
      sc1 = __builtin_amdgcn_wmma_f32_16x16x32_bf16(false, qf[c].v, false, k1.v, (short)0, sc1, false, false);
    }
    // online softmax: row r in this lane is query row (r + 8*h2); keys live across 16 lanes
    float cm[8];
    #pragma unroll
    for (int r = 0; r < 8; ++r){
      float mx = fmaxf(sc0[r], sc1[r]);
      #pragma unroll
      for (int o = 8; o >= 1; o >>= 1) mx = fmaxf(mx, __shfl_xor(mx, o, 32));
      float mnew = fmaxf(rowm[r], mx);
      cm[r] = __expf(rowm[r] - mnew);
      rowm[r] = mnew;
      float p0 = __expf(sc0[r] - mnew), p1 = __expf(sc1[r] - mnew);
      sc0[r] = p0; sc1[r] = p1;
      float sum = p0 + p1;
      #pragma unroll
      for (int o = 8; o >= 1; o >>= 1) sum += __shfl_xor(sum, o, 32);
      rowl[r] = rowl[r]*cm[r] + sum;
    }
    #pragma unroll
    for (int t = 0; t < 5; ++t)
      #pragma unroll
      for (int r = 0; r < 8; ++r) acc[t][r] *= cm[r];

    // transpose P (C-layout) -> A-fragment layout via LDS
    __syncthreads();
    #pragma unroll
    for (int r = 0; r < 8; ++r){
      pTile[(r + 8*h2)*32 + lm]      = f2bf(sc0[r]);
      pTile[(r + 8*h2)*32 + 16 + lm] = f2bf(sc1[r]);
    }
    __syncthreads();
    FragB pf;
    const unsigned int* pt = (const unsigned int*)pTile + lm*16;
    #pragma unroll
    for (int r = 0; r < 8; ++r) pf.d[r] = pt[fragDw(r,h2)];

    // O += P(16x32) x V(32 x 80): V^T rows contiguous along keys
    #pragma unroll
    for (int t = 0; t < 5; ++t){
      FragB vf;
      const unsigned int* vr = (const unsigned int*)(V + ((size_t)bh*DPV + t*16 + lm)*SS + kk);
      #pragma unroll
      for (int r = 0; r < 8; ++r) vf.d[r] = vr[fragDw(r,h2)];
      acc[t] = __builtin_amdgcn_wmma_f32_16x16x32_bf16(false, pf.v, false, vf.v, (short)0, acc[t], false, false);
    }
  }

  // normalize and store AO[b, s, h*72 + d] (bf16), skipping the d >= 72 padding
  #pragma unroll
  for (int t = 0; t < 5; ++t){
    int d = t*16 + lm;
    if (d < DD){
      #pragma unroll
      for (int r = 0; r < 8; ++r){
        int srow = s0 + r + 8*h2;
        AO[((size_t)(b*SS + srow))*EE + h*DD + d] = f2bf(acc[t][r] / rowl[r]);
      }
    }
  }
}

// ---------------- workspace layout ----------------
static const size_t SZ_XB  = (size_t)MTOT*EE*2;          // 18,874,368
static const size_t SZ_W   = (size_t)EE*EE*2;            //  2,654,208
static const size_t SZ_QP  = (size_t)BB*HH*SS*DPQK*2;    // 25,165,824
static const size_t SZ_VT  = (size_t)BB*HH*DPV*SS*2;     // 20,971,520
static const size_t SZ_AO  = (size_t)MTOT*EE*2;

static const size_t OFF_XB = 0;
static const size_t OFF_WQ = OFF_XB + SZ_XB;
static const size_t OFF_WK = OFF_WQ + SZ_W;
static const size_t OFF_WV = OFF_WK + SZ_W;
static const size_t OFF_WO = OFF_WV + SZ_W;
static const size_t OFF_QP = OFF_WO + SZ_W;
static const size_t OFF_KP = OFF_QP + SZ_QP;
static const size_t OFF_VT = OFF_KP + SZ_QP;
static const size_t OFF_AO = OFF_VT + SZ_VT;             // total ~120 MB

extern "C" void kernel_launch(void* const* d_in, const int* in_sizes, int n_in,
                              void* d_out, int out_size, void* d_ws, size_t ws_size,
                              hipStream_t stream){
  (void)in_sizes; (void)n_in; (void)out_size; (void)ws_size;
  const float* hX = (const float*)d_in[0];
  const float* Wq = (const float*)d_in[1];
  const float* bq = (const float*)d_in[2];
  const float* Wk = (const float*)d_in[3];
  const float* bk = (const float*)d_in[4];
  const float* Wv = (const float*)d_in[5];
  const float* bv = (const float*)d_in[6];
  const float* Wo = (const float*)d_in[7];
  const float* bo = (const float*)d_in[8];

  char* ws = (char*)d_ws;
  unsigned short* Xb  = (unsigned short*)(ws + OFF_XB);
  unsigned short* Wqb = (unsigned short*)(ws + OFF_WQ);
  unsigned short* Wkb = (unsigned short*)(ws + OFF_WK);
  unsigned short* Wvb = (unsigned short*)(ws + OFF_WV);
  unsigned short* Wob = (unsigned short*)(ws + OFF_WO);
  unsigned short* Qp  = (unsigned short*)(ws + OFF_QP);
  unsigned short* Kp  = (unsigned short*)(ws + OFF_KP);
  unsigned short* Vt  = (unsigned short*)(ws + OFF_VT);
  unsigned short* AO  = (unsigned short*)(ws + OFF_AO);

  // 1) fp32 -> bf16 conversions
  int nX = MTOT*EE;
  k_f32_to_bf16<<<(nX+255)/256, 256, 0, stream>>>(hX, Xb, nX);
  int nW = EE*EE;
  k_f32_to_bf16<<<(nW+255)/256, 256, 0, stream>>>(Wq, Wqb, nW);
  k_f32_to_bf16<<<(nW+255)/256, 256, 0, stream>>>(Wk, Wkb, nW);
  k_f32_to_bf16<<<(nW+255)/256, 256, 0, stream>>>(Wv, Wvb, nW);
  k_f32_to_bf16<<<(nW+255)/256, 256, 0, stream>>>(Wo, Wob, nW);

  // 2) zero Qp+Kp+Vt (contiguous) so D-padding is real zeros
  int nz = (int)((SZ_QP*2 + SZ_VT) / 4);
  k_zero_u32<<<(nz+255)/256, 256, 0, stream>>>((unsigned int*)(ws + OFF_QP), nz);

  // 3) projections (register-blocked WMMA GEMMs, 64x64 per wave)
  dim3 g(MTOT/64, EE/64);
  k_gemm<<<g, 32, 0, stream>>>(Xb, Wqb, bq, Qp, MODE_PAD);
  k_gemm<<<g, 32, 0, stream>>>(Xb, Wkb, bk, Kp, MODE_PAD);
  k_gemm<<<g, 32, 0, stream>>>(Xb, Wvb, bv, Vt, MODE_VT);

  // 4) RoPE (scale 1/sqrt(72) folded into Q)
  int nr = BB*HH*SS*36;
  k_rope<<<(nr+255)/256, 256, 0, stream>>>(Qp, 0.11785113019775793f, nr);
  k_rope<<<(nr+255)/256, 256, 0, stream>>>(Kp, 1.0f, nr);

  // 5) flash attention
  k_attn<<<BB*HH*(SS/16), 32, 0, stream>>>(Qp, Kp, Vt, AO);

  // 6) output projection, fp32 out
  k_gemm<<<g, 32, 0, stream>>>(AO, Wob, bo, d_out, MODE_F32);
}
